// TransformerXL_71528385348090
// MI455X (gfx1250) — matrix operand order repaired and checked by reading, not compiled, not run on hardware
//
#include <hip/hip_runtime.h>

// ---------------------------------------------------------------------------
// TransformerXL forward loss on MI455X (gfx1250, wave32, WMMA bf16)
// ---------------------------------------------------------------------------

typedef __bf16 bf16;
typedef __attribute__((ext_vector_type(16))) __bf16 v16bf;
typedef __attribute__((ext_vector_type(8)))  float  v8f;
typedef __attribute__((ext_vector_type(4)))  int    v4i;

constexpr int QLEN   = 512;
constexpr int MLEN   = 512;
constexpr int KLEN   = 1024;   // MLEN + QLEN
constexpr int BSZ    = 4;
constexpr int NH     = 16;
constexpr int DH     = 64;
constexpr int DM     = 1024;   // NH*DH
constexpr int DI     = 4096;
constexpr int NTOK   = 32000;
constexpr int NLAYER = 6;
constexpr int NBN    = BSZ * NH;      // 64 attention batches
constexpr int NTOKNS = QLEN * BSZ;    // 2048 query tokens

// GEMM tiling: 256 threads = 8 waves, block tile 128x64, each wave 32x32.
constexpr int BM = 128;
constexpr int BN = 64;
constexpr int BK = 32;
constexpr int KP = 40;  // padded LDS row stride in bf16 (80B: 16B aligned, conflict-free)

// ---------------------------------------------------------------------------
// gfx1250 async global->LDS staging (ASYNCcnt path), guarded so the code
// degrades to the proven VGPR-bounce path if the builtin is not declared.
// Prototype (from hipcc diagnostics): (int4 AS1*, int4 AS3*, imm i32, imm i32)
// ---------------------------------------------------------------------------
#define AS1 __attribute__((address_space(1)))
#define AS3 __attribute__((address_space(3)))

#if __has_builtin(__builtin_amdgcn_global_load_async_to_lds_b128)
#define USE_ASYNC_LDS 1
#else
#define USE_ASYNC_LDS 0
#endif

__device__ __forceinline__ void stage_b128(const bf16* __restrict__ g, bf16* l) {
#if USE_ASYNC_LDS
  __builtin_amdgcn_global_load_async_to_lds_b128(
      (AS1 v4i*)(bf16*)g, (AS3 v4i*)l, 0, 0);
#else
  *(uint4*)l = *(const uint4*)g;
#endif
}

__device__ __forceinline__ void stage_wait() {
#if USE_ASYNC_LDS
#if __has_builtin(__builtin_amdgcn_s_wait_asynccnt)
  __builtin_amdgcn_s_wait_asynccnt(0);
#else
  asm volatile("s_wait_asynccnt 0x0" ::: "memory");
#endif
#endif
}

// ---------------------------------------------------------------------------
// f32 -> bf16 conversion (weights, done once per launch).  NT loads: the f32
// weights are dead after this, keep them out of the 192MB L2 working set.
// ---------------------------------------------------------------------------
__global__ void cvt_f32_bf16_kernel(const float* __restrict__ in,
                                    bf16* __restrict__ out, long n) {
  long i = (long)blockIdx.x * blockDim.x + threadIdx.x;
  long stride = (long)gridDim.x * blockDim.x;
  for (; i < n; i += stride) out[i] = (bf16)__builtin_nontemporal_load(&in[i]);
}

// ---------------------------------------------------------------------------
// Generic batched bf16 GEMM, f32 accumulate via v_wmma_f32_16x16x32_bf16.
//   C[M,N] = A[M,K] * op(B)      op(B): BT=true  -> B is N x K row-major
//                                        BT=false -> B is K x N row-major
// Optional epilogue: +bias[N], relu, bf16 copy of C.
// ---------------------------------------------------------------------------
template<bool BT>
__global__ __launch_bounds__(256)
void gemm_bf16_kernel(const bf16* __restrict__ A, const bf16* __restrict__ B,
                      float* __restrict__ C, bf16* __restrict__ Cb,
                      const float* __restrict__ bias, int relu,
                      int M, int N, int K, int lda, int ldb, int ldc,
                      long sA, long sB, long sC) {
  __shared__ bf16 As[BM * KP];
  __shared__ bf16 Bs[BN * KP];

  const int z = blockIdx.z;
  A += (long)z * sA;  B += (long)z * sB;  C += (long)z * sC;
  if (Cb) Cb += (long)z * sC;

  const int m0 = blockIdx.y * BM, n0 = blockIdx.x * BN;
  const int tid  = threadIdx.x;
  const int lane = tid & 31, wave = tid >> 5;
  const int wm = wave >> 1, wn = wave & 1;      // 4 x 2 wave grid
  const int half = lane >> 4, l15 = lane & 15;

  const v8f vzero = {0.f, 0.f, 0.f, 0.f, 0.f, 0.f, 0.f, 0.f};
  v8f acc[2][2];
#pragma unroll
  for (int a = 0; a < 2; a++)
#pragma unroll
    for (int b = 0; b < 2; b++) acc[a][b] = vzero;

  for (int k0 = 0; k0 < K; k0 += BK) {
    // stage A tile (BM x BK): async b128 direct to LDS (no VGPR bounce)
    for (int idx = tid; idx < BM * (BK / 8); idx += 256) {
      int r = idx >> 2, c = (idx & 3) * 8;
      stage_b128(&A[(long)(m0 + r) * lda + k0 + c], &As[r * KP + c]);
    }
    // stage B tile as [BN][BK] (N-major so B fragments are contiguous-K)
    if (BT) {
      for (int idx = tid; idx < BN * (BK / 8); idx += 256) {
        int r = idx >> 2, c = (idx & 3) * 8;
        stage_b128(&B[(long)(n0 + r) * ldb + k0 + c], &Bs[r * KP + c]);
      }
    } else {
      for (int idx = tid; idx < BN * BK; idx += 256) {
        int k = idx / BN, n = idx % BN;            // consecutive tid -> coalesced global n
        Bs[n * KP + k] = B[(long)(k0 + k) * ldb + n0 + n];
      }
    }
    if (k0 + BK < K)  // prefetch next A tile -> global_prefetch_b8
      __builtin_prefetch(&A[(long)(m0 + (tid & (BM - 1))) * lda + k0 + BK], 0, 0);
    stage_wait();
    __syncthreads();

    // fragments: contiguous k = 16*half + e in both A and B (same order)
    v16bf af[2], bq[2];
#pragma unroll
    for (int a = 0; a < 2; a++) {
      const bf16* ap = &As[(wm * 32 + a * 16 + l15) * KP + 16 * half];
#pragma unroll
      for (int e = 0; e < 8; e++)  af[a][e] = ap[e];       // K = 16*half + e
#pragma unroll
      for (int e = 8; e < 16; e++) af[a][e] = ap[e];       // K = 16*half + e
    }
#pragma unroll
    for (int b = 0; b < 2; b++) {
      const bf16* bp = &Bs[(wn * 32 + b * 16 + l15) * KP + 16 * half];
#pragma unroll
      for (int e = 0; e < 16; e++) bq[b][e] = bp[e];       // K = 16*half + e
    }
#pragma unroll
    for (int a = 0; a < 2; a++)
#pragma unroll
      for (int b = 0; b < 2; b++)
        acc[a][b] = __builtin_amdgcn_wmma_f32_16x16x32_bf16(
            false, af[a], false, bq[b], (short)0, acc[a][b], false, false);
    __syncthreads();
  }

  // epilogue: C rows = base + v + 8*half, col = base + (lane&15)
#pragma unroll
  for (int a = 0; a < 2; a++)
#pragma unroll
    for (int b = 0; b < 2; b++) {
      int col = n0 + wn * 32 + b * 16 + l15;
      float bv = bias ? bias[col] : 0.f;
#pragma unroll
      for (int v = 0; v < 8; v++) {
        int row = m0 + wm * 32 + a * 16 + v + 8 * half;
        float x = acc[a][b][v] + bv;
        if (relu) x = fmaxf(x, 0.f);
        C[(long)row * ldc + col] = x;
        if (Cb) Cb[(long)row * ldc + col] = (bf16)x;
      }
    }
}

// ---------------------------------------------------------------------------
// Embedding lookup: h = emb[data] * sqrt(DM)
// ---------------------------------------------------------------------------
__global__ void embed_kernel(const int* __restrict__ data, const float* __restrict__ emb,
                             float* __restrict__ hf, bf16* __restrict__ hb) {
  long idx = (long)blockIdx.x * 256 + threadIdx.x;   // NTOKNS*DM threads
  int t = (int)(idx >> 10), d = (int)(idx & 1023);
  float v = emb[(long)data[t] * DM + d] * 32.0f;     // sqrt(1024)
  hf[idx] = v;  hb[idx] = (bf16)v;
}

// ---------------------------------------------------------------------------
// Sinusoidal relative positions (descending klen-1..0), bf16
// ---------------------------------------------------------------------------
__global__ void posemb_kernel(bf16* __restrict__ posb) {
  long idx = (long)blockIdx.x * 256 + threadIdx.x;   // KLEN*DM threads
  int j = (int)(idx >> 10), d = (int)(idx & 1023);
  int pos = (KLEN - 1) - j;
  int dd = d & 511;
  float inv = __expf(-9.210340371976184f * (float)dd * (1.0f / 512.0f)); // 10000^{-dd/512}
  float a = (float)pos * inv;
  posb[idx] = (bf16)((d < 512) ? __sinf(a) : __cosf(a));
}

// ---------------------------------------------------------------------------
// cat = [mems[layer]; h]  ->  bf16, token-major (klen*bsz, DM)
// ---------------------------------------------------------------------------
__global__ void concat_kernel(const float* __restrict__ memsL, const float* __restrict__ hf,
                              bf16* __restrict__ catb) {
  long idx = (long)blockIdx.x * 256 + threadIdx.x;   // KLEN*BSZ*DM threads
  int t = (int)(idx >> 10), d = (int)(idx & 1023);
  int k = t >> 2, b = t & 3;
  float v = (k < MLEN) ? memsL[idx]
                       : hf[(long)((k - MLEN) * BSZ + b) * DM + d];
  catb[idx] = (bf16)v;
}

// ---------------------------------------------------------------------------
// Re-layout q/k/v from heads; add r_w/r_r biases to q.
//   qw,qr: (bn, QLEN, DH)   kT: (bn, DH, KLEN)   vB: (bn, KLEN, DH)
// ---------------------------------------------------------------------------
__global__ void prep_attn_kernel(const float* __restrict__ heads,
                                 const float* __restrict__ rwb, const float* __restrict__ rrb,
                                 bf16* __restrict__ qw, bf16* __restrict__ qr,
                                 bf16* __restrict__ kT, bf16* __restrict__ vB) {
  long idx = (long)blockIdx.x * 256 + threadIdx.x;   // NBN*KLEN*DH threads
  int d  = (int)(idx & 63);
  int j  = (int)((idx >> 6) & 1023);
  int bn = (int)(idx >> 16);
  int n = bn & 15, b = bn >> 4;
  const float* hd = heads + ((long)j * BSZ + b) * (3 * DM);
  int e = n * DH + d;
  kT[((long)bn * DH + d) * KLEN + j]  = (bf16)hd[DM + e];
  vB[((long)bn * KLEN + j) * DH + d]  = (bf16)hd[2 * DM + e];
  if (j >= MLEN) {
    long qi = ((long)bn * QLEN + (j - MLEN)) * DH + d;
    float qv = hd[e];
    qw[qi] = (bf16)(qv + rwb[e]);
    qr[qi] = (bf16)(qv + rrb[e]);
  }
}

// r (KLEN, DM) f32 -> rT (bn, DH, KLEN) bf16 replicated over b
__global__ void rprep_kernel(const float* __restrict__ rf, bf16* __restrict__ rT) {
  long idx = (long)blockIdx.x * 256 + threadIdx.x;   // KLEN*DM threads
  int e = (int)(idx & 1023), j = (int)(idx >> 10);
  int n = e >> 6, d = e & 63;
  bf16 v = (bf16)rf[idx];
#pragma unroll
  for (int b = 0; b < BSZ; b++)
    rT[(((long)(b * NH + n)) * DH + d) * KLEN + j] = v;
}

// ---------------------------------------------------------------------------
// softmax over keys with analytic rel-shift:  BD[i,j] = BDp[i, j+511-i]
// mask j > i+MLEN.  prob written as bf16 (masked -> 0).
// ---------------------------------------------------------------------------
__global__ __launch_bounds__(256)
void attn_softmax_kernel(const float* __restrict__ AC, const float* __restrict__ BD,
                         bf16* __restrict__ prob) {
  __shared__ float red[256];
  const int i = blockIdx.x;
  const long base = ((long)blockIdx.y * QLEN + i) * KLEN;
  const float* ac = AC + base;
  const float* bd = BD + base;
  bf16* pr = prob + base;
  const int tid = threadIdx.x;
  const int jmax = i + MLEN;

  float sc[4], lm = -3.0e38f;
#pragma unroll
  for (int q = 0; q < 4; q++) {
    int j = tid + q * 256;
    sc[q] = (j <= jmax) ? (ac[j] + bd[j + (QLEN - 1) - i]) * 0.125f : -3.0e38f;
    lm = fmaxf(lm, sc[q]);
  }
  red[tid] = lm; __syncthreads();
  for (int o = 128; o > 0; o >>= 1) { if (tid < o) red[tid] = fmaxf(red[tid], red[tid + o]); __syncthreads(); }
  float mx = red[0]; __syncthreads();

  float e[4], ls = 0.f;
#pragma unroll
  for (int q = 0; q < 4; q++) { e[q] = __expf(sc[q] - mx); ls += e[q]; }
  red[tid] = ls; __syncthreads();
  for (int o = 128; o > 0; o >>= 1) { if (tid < o) red[tid] += red[tid + o]; __syncthreads(); }
  float inv = 1.0f / red[0];
#pragma unroll
  for (int q = 0; q < 4; q++) {
    int j = tid + q * 256;
    pr[j] = (bf16)((j <= jmax) ? e[q] * inv : 0.0f);
  }
}

// attn (bn, QLEN, DH) f32 -> token-major (NTOKNS, DM) bf16
__global__ void gather_attn_kernel(const float* __restrict__ attn, bf16* __restrict__ av) {
  long idx = (long)blockIdx.x * 256 + threadIdx.x;   // NTOKNS*DM threads
  int e = (int)(idx & 1023), t = (int)(idx >> 10);
  int i = t >> 2, b = t & 3, n = e >> 6, d = e & 63;
  av[idx] = (bf16)attn[(((long)(b * NH + n)) * QLEN + i) * DH + d];
}

// ---------------------------------------------------------------------------
// h = LayerNorm(h + delta) (in place) + bf16 copy
// ---------------------------------------------------------------------------
__global__ __launch_bounds__(256)
void add_ln_kernel(float* __restrict__ h, const float* __restrict__ delta,
                   const float* __restrict__ gam, const float* __restrict__ bet,
                   bf16* __restrict__ hb) {
  __shared__ float red[256];
  const long base = (long)blockIdx.x * DM;
  const int tid = threadIdx.x;
  float x[4], s = 0.f;
#pragma unroll
  for (int q = 0; q < 4; q++) { int d = tid + q * 256; x[q] = h[base + d] + delta[base + d]; s += x[q]; }
  red[tid] = s; __syncthreads();
  for (int o = 128; o > 0; o >>= 1) { if (tid < o) red[tid] += red[tid + o]; __syncthreads(); }
  float mu = red[0] * (1.0f / DM); __syncthreads();
  float vs = 0.f;
#pragma unroll
  for (int q = 0; q < 4; q++) { float d0 = x[q] - mu; vs += d0 * d0; }
  red[tid] = vs; __syncthreads();
  for (int o = 128; o > 0; o >>= 1) { if (tid < o) red[tid] += red[tid + o]; __syncthreads(); }
  float rstd = rsqrtf(red[0] * (1.0f / DM) + 1e-5f);
#pragma unroll
  for (int q = 0; q < 4; q++) {
    int d = tid + q * 256;
    float y = (x[q] - mu) * rstd * gam[d] + bet[d];
    h[base + d] = y;  hb[base + d] = (bf16)y;
  }
}

// ---------------------------------------------------------------------------
// Final loss: streamed logits + online log-softmax.  One block = 16 token
// rows; 8 waves stride over 2000 vocab tiles (16 wide), WMMA K-loop of 32.
// 65MB bf16 vocab matrix stays L2-resident (192MB L2).
// ---------------------------------------------------------------------------
__global__ __launch_bounds__(256)
void loss_kernel(const bf16* __restrict__ hb, const bf16* __restrict__ embb,
                 const float* __restrict__ obias, const int* __restrict__ target,
                 float* __restrict__ loss) {
  __shared__ bf16  Ah[16 * 1032];
  __shared__ float redm[8][16];
  __shared__ float reds[8][16];
  __shared__ float tred[16][17];

  const int tid = threadIdx.x, lane = tid & 31, wave = tid >> 5;
  const int half = lane >> 4, l15 = lane & 15;
  const int t0 = blockIdx.x * 16;

  // stage 16 x 1024 bf16 activation tile
  for (int idx = tid; idx < 16 * 128; idx += 256) {
    int r = idx >> 7, c = (idx & 127) * 8;
    stage_b128(&hb[(long)(t0 + r) * DM + c], &Ah[r * 1032 + c]);
  }
  stage_wait();
  __syncthreads();

  const v8f vzero = {0.f, 0.f, 0.f, 0.f, 0.f, 0.f, 0.f, 0.f};
  float m[8], s[8];
#pragma unroll
  for (int v = 0; v < 8; v++) { m[v] = -3.0e38f; s[v] = 0.f; }

  for (int nt = wave; nt < NTOK / 16; nt += 8) {
    const int n0v = nt * 16;
    v8f acc = vzero;
    const bf16* brow = embb + (long)(n0v + l15) * DM + 16 * half;
    for (int kt = 0; kt < DM / 32; kt++) {
      v16bf afr, bfr;
      const bf16* ap = &Ah[l15 * 1032 + kt * 32 + 16 * half];
#pragma unroll
      for (int e = 0; e < 8; e++)  afr[e] = ap[e];
#pragma unroll
      for (int e = 8; e < 16; e++) afr[e] = ap[e];
      const bf16* bp = brow + kt * 32;
#pragma unroll
      for (int e = 0; e < 16; e++) bfr[e] = bp[e];
      acc = __builtin_amdgcn_wmma_f32_16x16x32_bf16(
          false, afr, false, bfr, (short)0, acc, false, false);
    }
    // online softmax update; each row's 16 cols live in a 16-lane group
#pragma unroll
    for (int v = 0; v < 8; v++) {
      float x = acc[v] + obias[n0v + l15];
      float tm = x;
      for (int o = 1; o < 16; o <<= 1) tm = fmaxf(tm, __shfl_xor(tm, o, 32));
      float mn = fmaxf(m[v], tm);
      float ts = __expf(x - mn);
      for (int o = 1; o < 16; o <<= 1) ts += __shfl_xor(ts, o, 32);
      s[v] = s[v] * __expf(m[v] - mn) + ts;
      m[v] = mn;
    }
  }
  if (l15 == 0) {
#pragma unroll
    for (int v = 0; v < 8; v++) { int r = v + 8 * half; redm[wave][r] = m[v]; reds[wave][r] = s[v]; }
  }

  // target logit: 16 threads per row, 64 elems each
  {
    int r = tid >> 4, c = tid & 15;
    int tgt = target[t0 + r];
    const bf16* er = embb + (long)tgt * DM + c * 64;
    const bf16* hr = &Ah[r * 1032 + c * 64];
    float a2 = 0.f;
    for (int d = 0; d < 64; d++) a2 += (float)hr[d] * (float)er[d];
    tred[r][c] = a2;
  }
  __syncthreads();

  if (tid < 16) {
    int r = tid;
    float dot = 0.f;
    for (int c = 0; c < 16; c++) dot += tred[r][c];
    dot += obias[target[t0 + r]];
    float M = -3.0e38f;
    for (int w = 0; w < 8; w++) M = fmaxf(M, redm[w][r]);
    float S = 0.f;
    for (int w = 0; w < 8; w++) S += reds[w][r] * __expf(redm[w][r] - M);
    loss[t0 + r] = (M + __logf(S)) - dot;   // -(logit_t - lse)
  }
}

// ---------------------------------------------------------------------------
// Host orchestration
// ---------------------------------------------------------------------------
extern "C" void kernel_launch(void* const* d_in, const int* in_sizes, int n_in,
                              void* d_out, int out_size, void* d_ws, size_t ws_size,
                              hipStream_t stream) {
  (void)in_sizes; (void)n_in; (void)out_size; (void)ws_size;
  const int*   data   = (const int*)  d_in[0];
  const int*   target = (const int*)  d_in[1];
  const float* mems   = (const float*)d_in[2];
  const float* emb    = (const float*)d_in[3];
  const float* obias  = (const float*)d_in[4];
  const float* rwb    = (const float*)d_in[5];
  const float* rrb    = (const float*)d_in[6];
  const float* qkv_w  = (const float*)d_in[7];
  const float* r_w    = (const float*)d_in[8];
  const float* o_w    = (const float*)d_in[9];
  const float* ln1g   = (const float*)d_in[10];
  const float* ln1b   = (const float*)d_in[11];
  const float* ff1w   = (const float*)d_in[12];
  const float* ff1b   = (const float*)d_in[13];
  const float* ff2w   = (const float*)d_in[14];
  const float* ff2b   = (const float*)d_in[15];
  const float* ln2g   = (const float*)d_in[16];
  const float* ln2b   = (const float*)d_in[17];
  float* loss = (float*)d_out;

  size_t off = 0;
  auto alloc = [&](size_t nbytes) -> void* {
    void* p = (void*)((char*)d_ws + off);
    off += (nbytes + 255) & ~(size_t)255;
    return p;
  };
  // bf16 buffers
  bf16* Wqkv = (bf16*)alloc((size_t)NLAYER * 3 * DM * DM * sizeof(bf16));
  bf16* Wr   = (bf16*)alloc((size_t)NLAYER * DM * DM * sizeof(bf16));
  bf16* Wo   = (bf16*)alloc((size_t)NLAYER * DM * DM * sizeof(bf16));
  bf16* Wf1  = (bf16*)alloc((size_t)NLAYER * DI * DM * sizeof(bf16));
  bf16* Wf2  = (bf16*)alloc((size_t)NLAYER * DM * DI * sizeof(bf16));
  bf16* Wemb = (bf16*)alloc((size_t)NTOK * DM * sizeof(bf16));
  bf16* hb    = (bf16*)alloc((size_t)NTOKNS * DM * sizeof(bf16));
  bf16* posb  = (bf16*)alloc((size_t)KLEN * DM * sizeof(bf16));
  bf16* catb  = (bf16*)alloc((size_t)KLEN * BSZ * DM * sizeof(bf16));
  bf16* qwb   = (bf16*)alloc((size_t)NBN * QLEN * DH * sizeof(bf16));
  bf16* qrb   = (bf16*)alloc((size_t)NBN * QLEN * DH * sizeof(bf16));
  bf16* kTb   = (bf16*)alloc((size_t)NBN * DH * KLEN * sizeof(bf16));
  bf16* vBb   = (bf16*)alloc((size_t)NBN * KLEN * DH * sizeof(bf16));
  bf16* rTb   = (bf16*)alloc((size_t)NBN * DH * KLEN * sizeof(bf16));
  bf16* probb = (bf16*)alloc((size_t)NBN * QLEN * KLEN * sizeof(bf16));
  bf16* avb   = (bf16*)alloc((size_t)NTOKNS * DM * sizeof(bf16));
  bf16* ffhb  = (bf16*)alloc((size_t)NTOKNS * DI * sizeof(bf16));
  // f32 buffers
  float* hf     = (float*)alloc((size_t)NTOKNS * DM * sizeof(float));
  float* headsf = (float*)alloc((size_t)KLEN * BSZ * 3 * DM * sizeof(float));
  float* rf     = (float*)alloc((size_t)KLEN * DM * sizeof(float));
  float* ACf    = (float*)alloc((size_t)NBN * QLEN * KLEN * sizeof(float));
  float* BDf    = (float*)alloc((size_t)NBN * QLEN * KLEN * sizeof(float));
  float* attnf  = (float*)alloc((size_t)NBN * QLEN * DH * sizeof(float));
  float* aoutf  = (float*)alloc((size_t)NTOKNS * DM * sizeof(float));
  float* ffhf   = (float*)alloc((size_t)NTOKNS * DI * sizeof(float));
  float* fff    = (float*)alloc((size_t)NTOKNS * DM * sizeof(float));

  auto cvt = [&](const float* s, bf16* d, long n) {
    long g = (n + 2047) / 2048; if (g > 16384) g = 16384;
    cvt_f32_bf16_kernel<<<dim3((unsigned)g), dim3(256), 0, stream>>>(s, d, n);
  };
  auto gemm_bt = [&](const bf16* A, const bf16* B, float* C, bf16* Cb,
                     const float* bias, int relu, int M, int N, int K,
                     int lda, int ldb, int ldc, long sA, long sB, long sC, int batch) {
    dim3 g((unsigned)(N / BN), (unsigned)(M / BM), (unsigned)batch);
    gemm_bf16_kernel<true><<<g, dim3(256), 0, stream>>>(A, B, C, Cb, bias, relu,
                                                        M, N, K, lda, ldb, ldc, sA, sB, sC);
  };
  auto gemm_bn = [&](const bf16* A, const bf16* B, float* C, bf16* Cb,
                     const float* bias, int relu, int M, int N, int K,
                     int lda, int ldb, int ldc, long sA, long sB, long sC, int batch) {
    dim3 g((unsigned)(N / BN), (unsigned)(M / BM), (unsigned)batch);
    gemm_bf16_kernel<false><<<g, dim3(256), 0, stream>>>(A, B, C, Cb, bias, relu,
                                                         M, N, K, lda, ldb, ldc, sA, sB, sC);
  };

  // ---- one-time conversions ----
  cvt(qkv_w, Wqkv, (long)NLAYER * 3 * DM * DM);
  cvt(r_w,   Wr,   (long)NLAYER * DM * DM);
  cvt(o_w,   Wo,   (long)NLAYER * DM * DM);
  cvt(ff1w,  Wf1,  (long)NLAYER * DI * DM);
  cvt(ff2w,  Wf2,  (long)NLAYER * DM * DI);
  cvt(emb,   Wemb, (long)NTOK * DM);

  embed_kernel <<<dim3(NTOKNS * DM / 256), dim3(256), 0, stream>>>(data, emb, hf, hb);
  posemb_kernel<<<dim3(KLEN * DM / 256),   dim3(256), 0, stream>>>(posb);

  // ---- transformer layers ----
  for (int l = 0; l < NLAYER; l++) {
    const float* memsL = mems + (long)l * MLEN * BSZ * DM;
    concat_kernel<<<dim3(KLEN * BSZ * DM / 256), dim3(256), 0, stream>>>(memsL, hf, catb);

    // heads = cat @ qkv_w^T : (4096 x 3072 x 1024)
    gemm_bt(catb, Wqkv + (long)l * 3 * DM * DM, headsf, nullptr, nullptr, 0,
            KLEN * BSZ, 3 * DM, DM, DM, DM, 3 * DM, 0, 0, 0, 1);
    // r = pos_emb @ r_w^T : (1024 x 1024 x 1024)
    gemm_bt(posb, Wr + (long)l * DM * DM, rf, nullptr, nullptr, 0,
            KLEN, DM, DM, DM, DM, DM, 0, 0, 0, 1);

    prep_attn_kernel<<<dim3(NBN * KLEN * DH / 256), dim3(256), 0, stream>>>(
        headsf, rwb, rrb, qwb, qrb, kTb, vBb);
    rprep_kernel<<<dim3(KLEN * DM / 256), dim3(256), 0, stream>>>(rf, rTb);

    // AC[bn] = (q + r_w_bias) @ k^T : batched (512 x 1024 x 64)
    gemm_bn(qwb, kTb, ACf, nullptr, nullptr, 0, QLEN, KLEN, DH,
            DH, KLEN, KLEN, (long)QLEN * DH, (long)DH * KLEN, (long)QLEN * KLEN, NBN);
    // BD_pre[bn] = (q + r_r_bias) @ r^T : batched (512 x 1024 x 64)
    gemm_bn(qrb, rTb, BDf, nullptr, nullptr, 0, QLEN, KLEN, DH,
            DH, KLEN, KLEN, (long)QLEN * DH, (long)DH * KLEN, (long)QLEN * KLEN, NBN);

    attn_softmax_kernel<<<dim3(QLEN, NBN), dim3(256), 0, stream>>>(ACf, BDf, probb);

    // attn[bn] = prob @ v : batched (512 x 64 x 1024)
    gemm_bn(probb, vBb, attnf, nullptr, nullptr, 0, QLEN, DH, KLEN,
            KLEN, DH, DH, (long)QLEN * KLEN, (long)KLEN * DH, (long)QLEN * DH, NBN);

    gather_attn_kernel<<<dim3(NTOKNS * DM / 256), dim3(256), 0, stream>>>(attnf, avb);

    // attn_out = attn_vec @ o_w^T : (2048 x 1024 x 1024)
    gemm_bt(avb, Wo + (long)l * DM * DM, aoutf, nullptr, nullptr, 0,
            NTOKNS, DM, DM, DM, DM, DM, 0, 0, 0, 1);

    add_ln_kernel<<<dim3(NTOKNS), dim3(256), 0, stream>>>(
        hf, aoutf, ln1g + (long)l * DM, ln1b + (long)l * DM, hb);

    // ff_h = relu(h @ ff1^T + b1) : (2048 x 4096 x 1024), fused bias+relu+bf16
    gemm_bt(hb, Wf1 + (long)l * DI * DM, ffhf, ffhb, ff1b + (long)l * DI, 1,
            NTOKNS, DI, DM, DM, DM, DI, 0, 0, 0, 1);
    // ff = ff_h @ ff2^T + b2 : (2048 x 1024 x 4096)
    gemm_bt(ffhb, Wf2 + (long)l * DM * DI, fff, nullptr, ff2b + (long)l * DM, 0,
            NTOKNS, DM, DI, DI, DI, DM, 0, 0, 0, 1);

    add_ln_kernel<<<dim3(NTOKNS), dim3(256), 0, stream>>>(
        hf, fff, ln2g + (long)l * DM, ln2b + (long)l * DM, hb);
  }

  // ---- streamed log-softmax NLL over 32000-way vocab ----
  loss_kernel<<<dim3(NTOKNS / 16), dim3(256), 0, stream>>>(hb, Wemb, obias, target, loss);
}
